// GraphAttUnit_85048942396077
// MI455X (gfx1250) — matrix operand order repaired and checked by reading, not compiled, not run on hardware
//
#include <hip/hip_runtime.h>

// ---------------------------------------------------------------------------
// GraphAttUnit fused attention for gfx1250 (MI455X), bf16 WMMA path + TDM
// double-buffered async staging. B=8, C=512, H=W=64 -> N=4096, INTER=D=256.
// ---------------------------------------------------------------------------

typedef __attribute__((ext_vector_type(16))) __bf16 v16bf;
typedef __attribute__((ext_vector_type(8)))  float  v8f;
typedef __attribute__((ext_vector_type(4)))  unsigned int u32x4;
typedef __attribute__((ext_vector_type(8)))  int i32x8;
typedef __attribute__((ext_vector_type(4)))  int i32x4;

constexpr int Bsz  = 8;
constexpr int Cdim = 512;
constexpr int Ndim = 4096;
constexpr int Ddim = 256;
constexpr int NCHUNK = Ndim / 32;          // 128 column chunks

// TDM availability: device pass only; arity differs per toolchain.
#if defined(__gfx1250__) && __has_builtin(__builtin_amdgcn_tensor_load_to_lds)
#define USE_TDM 1
#if __has_include(<hip/amd_detail/amd_gfx1250_TDM.h>)
#define TDM_SIX_ARGS 1
#endif
#endif

__device__ __forceinline__ v8f wmma_bf16(v16bf a, v16bf b, v8f c) {
    return __builtin_amdgcn_wmma_f32_16x16x32_bf16(
        /*neg_a=*/false, a, /*neg_b=*/false, b,
        /*c_mod=*/(short)0, c, /*reuse_a=*/false, /*reuse_b=*/false);
}

union FragU { v16bf v; u32x4 u[2]; };

// A-fragment (16x32, M=row per lane%16): lane<16 holds K 0..15,
// lane>=16 holds K 16..31  => chunks at base+(L<16?0:16) and +8.
__device__ __forceinline__ v16bf load_fragA(const __bf16* base, int L) {
    FragU f;
    const __bf16* p = base + ((L < 16) ? 0 : 16);
    f.u[0] = *(const u32x4*)p;
    f.u[1] = *(const u32x4*)(p + 8);
    return f.v;
}

// B-fragment (32x16, N=col per lane%16): lane half selects K 0..15 / 16..31,
// 16 contiguous K elements per lane => chunks at p and p+8.
__device__ __forceinline__ v16bf load_fragB(const __bf16* p) {
    FragU f;
    f.u[0] = *(const u32x4*)p;
    f.u[1] = *(const u32x4*)(p + 8);
    return f.v;
}

#ifdef USE_TDM
// Issue TDM: DMA a [32 x 256] bf16 tile (16 KB, full contiguous rows of a
// [N x 256] tensor) from global into LDS. ISA 08 §8.3/8.4 D# layout.
__device__ __forceinline__ void tdm_phi_load(const __bf16* gsrc,
                                             unsigned ldsAddr) {
    unsigned long long ga = (unsigned long long)(uintptr_t)gsrc;
    u32x4 g0;
    g0[0] = 1u;                                   // count=1, user D#
    g0[1] = ldsAddr;                              // lds_addr (bytes)
    g0[2] = (unsigned)(ga & 0xFFFFFFFFull);       // global_addr lo
    g0[3] = (unsigned)((ga >> 32) & 0x01FFFFFFu)  // global_addr hi
          | (2u << 30);                           // type = 2 (image)
    i32x8 g1;
    g1[0] = (int)(1u << 16);          // wg_mask=0, data_size=1 (2B)
    g1[1] = (int)(256u << 16);        // tensor_dim0 = 256 (lo16 in [63:48])
    g1[2] = (int)(4096u << 16);       // dim0 hi=0 | tensor_dim1 lo16
    g1[3] = (int)(256u << 16);        // dim1 hi=0 | tile_dim0 = 256
    g1[4] = 32;                       // tile_dim1 = 32, tile_dim2 = 0
    g1[5] = 256;                      // tensor_dim0_stride = 256
    g1[6] = 0;
    g1[7] = 0;
    i32x4 gz4 = {0, 0, 0, 0};
#ifdef TDM_SIX_ARGS
    i32x8 gz8 = {0, 0, 0, 0, 0, 0, 0, 0};
    __builtin_amdgcn_tensor_load_to_lds(g0, g1, gz4, gz4, gz8, 0);
#else
    __builtin_amdgcn_tensor_load_to_lds(g0, g1, gz4, gz4, 0);
#endif
}
#endif

// ---------------------------------------------------------------------------
// f32 -> bf16 conversion
// ---------------------------------------------------------------------------
__global__ void cvt_bf16_kernel(const float* __restrict__ src,
                                __bf16* __restrict__ dst, int n) {
    int i = blockIdx.x * blockDim.x + threadIdx.x;
    if (i < n) dst[i] = (__bf16)src[i];
}

// ---------------------------------------------------------------------------
// Projection: outT[b, n, o] = sum_c w[o,c] * x[b,c,n] + bias[o], bf16 output.
// Grid: (N/64, B), 128 threads (4 waves, 16 rows each).
// ---------------------------------------------------------------------------
__global__ void __launch_bounds__(128)
proj_kernel(const float* __restrict__ x,     // [B, C, N]
            const __bf16* __restrict__ wbf,  // [D, C] bf16
            const float* __restrict__ bias,  // [D]
            __bf16* __restrict__ outT)       // [B, N, D] bf16
{
    __shared__ __bf16 xs[64][40];       // x chunk transposed [n][c], padded
    __shared__ __bf16 ts[4][16][16];    // per-wave store staging

    const int b  = blockIdx.y;
    const int nb = blockIdx.x * 64;
    const int t  = threadIdx.x;
    const int wv = t >> 5;
    const int L  = t & 31;
    const int Ll = L & 15;

    const float* xb = x + (size_t)b * Cdim * Ndim;

    v8f acc[16];
#pragma unroll
    for (int i = 0; i < 16; ++i)
#pragma unroll
        for (int r = 0; r < 8; ++r) acc[i][r] = 0.0f;

    for (int c0 = 0; c0 < Cdim; c0 += 32) {
        __syncthreads();
        // stage x[c0..c0+31, nb..nb+63] transposed into LDS (bf16)
#pragma unroll
        for (int rep = 0; rep < 4; ++rep) {
            int f4 = t + 128 * rep;          // 0..511 float4 tiles
            int cc = f4 >> 4;                // 0..31
            int n4 = (f4 & 15) << 2;         // 0..60
            float4 v = *(const float4*)(xb + (size_t)(c0 + cc) * Ndim + nb + n4);
            xs[n4 + 0][cc] = (__bf16)v.x;
            xs[n4 + 1][cc] = (__bf16)v.y;
            xs[n4 + 2][cc] = (__bf16)v.z;
            xs[n4 + 3][cc] = (__bf16)v.w;
        }
        __syncthreads();

        const int ln = (wv << 4) + Ll;
        v16bf A = load_fragA(&xs[ln][0], L);

#pragma unroll
        for (int ot = 0; ot < 16; ++ot) {
            const __bf16* bp = wbf + (size_t)(ot * 16 + Ll) * Cdim + c0
                             + ((L < 16) ? 0 : 16);
            v16bf Bf = load_fragB(bp);
            acc[ot] = wmma_bf16(A, Bf, acc[ot]);
        }
    }

    // bias add, bf16 convert, transpose through LDS, vectorized store
    __bf16* outb = outT + (size_t)b * Ndim * Ddim;
    const int n0w = nb + (wv << 4);
#pragma unroll
    for (int ot = 0; ot < 16; ++ot) {
        float bv = bias[ot * 16 + Ll];
#pragma unroll
        for (int r = 0; r < 8; ++r) {
            float xv = acc[ot][r] + bv;
            ts[wv][r + ((L < 16) ? 0 : 8)][Ll] = (__bf16)xv;
        }
        __builtin_amdgcn_wave_barrier();   // LDS same-wave is in-order
        if (L < 16) {
            const __bf16* rp = &ts[wv][L][0];
            u32x4 d0 = *(const u32x4*)rp;
            u32x4 d1 = *(const u32x4*)(rp + 8);
            __bf16* op = outb + (size_t)(n0w + L) * Ddim + ot * 16;
            *(u32x4*)op       = d0;
            *(u32x4*)(op + 8) = d1;
        }
        __builtin_amdgcn_wave_barrier();
    }
}

// ---------------------------------------------------------------------------
// Fused flash attention:
//   S = thetaX @ phiT^T ; S *= mask(col); S==0 -> -1e9; S/=4;
//   online softmax; O += P @ G ; out[b,d,n] = O / l
// Grid: (N/64, B), 128 threads (4 waves, 16 query rows each).
// phi chunks staged by the Tensor Data Mover, double-buffered so the DMA of
// chunk j+1 overlaps the WMMA/softmax work on chunk j.
// ---------------------------------------------------------------------------
__global__ void __launch_bounds__(128)
attn_kernel(const __bf16* __restrict__ thetaX,  // [B,N,D] bf16
            const __bf16* __restrict__ phiT,    // [B,N,D] bf16
            const __bf16* __restrict__ gx,      // [B,D,N] bf16
            const float*  __restrict__ mask,    // [B,N]
            float* __restrict__ out)            // [B,D,N]
{
    __shared__ __bf16 phis[2][32 * 256]; // double-buffered phi chunk, 2x16 KB
    __shared__ __bf16 plds[4][16][40];   // per-wave P staging, padded

    const int b  = blockIdx.y;
    const int t  = threadIdx.x;
    const int wv = t >> 5;
    const int L  = t & 31;
    const int Ll = L & 15;
    const int q0 = blockIdx.x * 64 + wv * 16;

    const __bf16* thb = thetaX + (size_t)b * Ndim * Ddim;
    const __bf16* phb = phiT   + (size_t)b * Ndim * Ddim;
    const __bf16* gxb = gx     + (size_t)b * Ddim * Ndim;
    const float*  mkb = mask   + (size_t)b * Ndim;

    // Q fragments: 8 K-chunks of 32 covering D=256
    v16bf qf[8];
    const int qrow = q0 + Ll;
#pragma unroll
    for (int kc = 0; kc < 8; ++kc)
        qf[kc] = load_fragA(thb + (size_t)qrow * Ddim + kc * 32, L);

    v8f acc[16];
#pragma unroll
    for (int i = 0; i < 16; ++i)
#pragma unroll
        for (int r = 0; r < 8; ++r) acc[i][r] = 0.0f;
    float mrow[8], lrow[8];
#pragma unroll
    for (int r = 0; r < 8; ++r) { mrow[r] = -3.0e38f; lrow[r] = 0.0f; }

    constexpr float LOG2E = 1.4426950408889634f;

#ifdef USE_TDM
    const unsigned ldsA0 = (unsigned)(unsigned long long)(uintptr_t)&phis[0][0];
    const unsigned ldsA1 = (unsigned)(unsigned long long)(uintptr_t)&phis[1][0];
    // prologue: start DMA of chunk 0 into buffer 0
    if (wv == 0) tdm_phi_load(phb, ldsA0);
#endif

    for (int jc = 0; jc < NCHUNK; ++jc) {
        const int j0  = jc * 32;
        const int cur = jc & 1;

        __syncthreads();   // all waves done reading buffer cur (2 iters ago)

        // prefetch next chunk of G rows (global_prefetch_b8)
        {
            int jn = (j0 + 32) & (Ndim - 1);
            __builtin_prefetch(gxb + (size_t)(t * 2)     * Ndim + jn, 0, 1);
            __builtin_prefetch(gxb + (size_t)(t * 2 + 1) * Ndim + jn, 0, 1);
        }

#ifdef USE_TDM
        if (wv == 0) {
            if (jc + 1 < NCHUNK) {
                // kick off DMA of next chunk into the other buffer, then wait
                // for the older transfer only (TDM completes in order).
                tdm_phi_load(phb + (size_t)(j0 + 32) * Ddim,
                             cur ? ldsA0 : ldsA1);
                __builtin_amdgcn_s_wait_tensorcnt(1);
            } else {
                __builtin_amdgcn_s_wait_tensorcnt(0);
            }
        }
        __syncthreads();   // publish TDM-written buffer cur to all waves
#else
        // fallback: cooperative staging of phi chunk [32 x 256] bf16 (16 KB)
        {
            const u32x4* src = (const u32x4*)(phb + (size_t)j0 * Ddim);
            u32x4* dst = (u32x4*)&phis[cur][0];
#pragma unroll
            for (int rep = 0; rep < 8; ++rep)
                dst[t + 128 * rep] = src[t + 128 * rep];
        }
        __syncthreads();
#endif
        const __bf16* ph = &phis[cur][0];

        // S tiles (two 16x16 column tiles), K = 256
        v8f s0, s1;
#pragma unroll
        for (int r = 0; r < 8; ++r) { s0[r] = 0.0f; s1[r] = 0.0f; }
#pragma unroll
        for (int kc = 0; kc < 8; ++kc) {
            const __bf16* bp0 = ph + (size_t)Ll * 256 + kc * 32
                              + ((L < 16) ? 0 : 16);
            s0 = wmma_bf16(qf[kc], load_fragB(bp0), s0);
            s1 = wmma_bf16(qf[kc], load_fragB(bp0 + 16 * 256), s1);
        }

        // mask * S ; ==0 -> -1e9 ; / TEMPERATURE
        const float mk0 = mkb[j0 + Ll];
        const float mk1 = mkb[j0 + 16 + Ll];
#pragma unroll
        for (int r = 0; r < 8; ++r) {
            float x0 = s0[r] * mk0; x0 = (x0 == 0.0f) ? -1e9f : x0; s0[r] = x0 * 0.25f;
            float x1 = s1[r] * mk1; x1 = (x1 == 0.0f) ? -1e9f : x1; s1[r] = x1 * 0.25f;
        }

        // online softmax; each 16-lane half owns row r (lanes<16) / r+8 (>=16)
#pragma unroll
        for (int r = 0; r < 8; ++r) {
            float cm = fmaxf(s0[r], s1[r]);
#pragma unroll
            for (int mski = 1; mski < 16; mski <<= 1)
                cm = fmaxf(cm, __shfl_xor(cm, mski, 32));
            float nm    = fmaxf(mrow[r], cm);
            float alpha = __builtin_amdgcn_exp2f((mrow[r] - nm) * LOG2E);
            mrow[r] = nm;
            float p0 = __builtin_amdgcn_exp2f((s0[r] - nm) * LOG2E);
            float p1 = __builtin_amdgcn_exp2f((s1[r] - nm) * LOG2E);
            float rs = p0 + p1;
#pragma unroll
            for (int mski = 1; mski < 16; mski <<= 1)
                rs += __shfl_xor(rs, mski, 32);
            lrow[r] = lrow[r] * alpha + rs;
#pragma unroll
            for (int dt = 0; dt < 16; ++dt) acc[dt][r] *= alpha;
            // restage P (C layout) -> LDS row-major for A-fragment reload
            plds[wv][r + ((L < 16) ? 0 : 8)][Ll]      = (__bf16)p0;
            plds[wv][r + ((L < 16) ? 0 : 8)][16 + Ll] = (__bf16)p1;
        }
        __builtin_amdgcn_wave_barrier();

        v16bf PA = load_fragA(&plds[wv][Ll][0], L);

        // O += P @ G : 16 d-tiles, B-fragment columns are rows of gx[b,d,:]
#pragma unroll
        for (int dt = 0; dt < 16; ++dt) {
            const __bf16* gp = gxb + (size_t)(dt * 16 + Ll) * Ndim + j0
                             + ((L < 16) ? 0 : 16);
            acc[dt] = wmma_bf16(PA, load_fragB(gp), acc[dt]);
        }
        __builtin_amdgcn_wave_barrier();
    }

    // finalize: O / l ; store out[b,d, q0 + m]; each lane owns 8 consecutive
    // query rows at fixed d -> two b128 stores per tile.
    float inv[8];
#pragma unroll
    for (int r = 0; r < 8; ++r) inv[r] = 1.0f / lrow[r];
    float* ob = out + (size_t)b * Ddim * Ndim;
#pragma unroll
    for (int dt = 0; dt < 16; ++dt) {
        int d = dt * 16 + Ll;
        float4 v0, v1;
        v0.x = acc[dt][0] * inv[0]; v0.y = acc[dt][1] * inv[1];
        v0.z = acc[dt][2] * inv[2]; v0.w = acc[dt][3] * inv[3];
        v1.x = acc[dt][4] * inv[4]; v1.y = acc[dt][5] * inv[5];
        v1.z = acc[dt][6] * inv[6]; v1.w = acc[dt][7] * inv[7];
        float* op = ob + (size_t)d * Ndim + q0 + ((L < 16) ? 0 : 8);
        *(float4*)op       = v0;
        *(float4*)(op + 4) = v1;
    }
}

// ---------------------------------------------------------------------------
// Launch
// ---------------------------------------------------------------------------
extern "C" void kernel_launch(void* const* d_in, const int* in_sizes, int n_in,
                              void* d_out, int out_size, void* d_ws, size_t ws_size,
                              hipStream_t stream) {
    const float* x_query   = (const float*)d_in[0];
    const float* x_support = (const float*)d_in[1];
    const float* sup_emb   = (const float*)d_in[2];
    const float* sup_mask  = (const float*)d_in[3];
    const float* theta_w   = (const float*)d_in[4];
    const float* theta_b   = (const float*)d_in[5];
    const float* phi_w     = (const float*)d_in[6];
    const float* phi_b     = (const float*)d_in[7];
    float* out = (float*)d_out;

    // workspace layout (bf16): thetaX | phiT | gx | theta_w | phi_w
    __bf16* thetaX = (__bf16*)d_ws;
    __bf16* phiTp  = thetaX + (size_t)Bsz * Ndim * Ddim;
    __bf16* gxbf   = phiTp  + (size_t)Bsz * Ndim * Ddim;
    __bf16* twbf   = gxbf   + (size_t)Bsz * Ddim * Ndim;
    __bf16* pwbf   = twbf   + (size_t)Ddim * Cdim;

    const int nw = Ddim * Cdim;            // 131072
    const int ng = Bsz * Ddim * Ndim;      // 8388608
    cvt_bf16_kernel<<<(nw + 255) / 256, 256, 0, stream>>>(theta_w, twbf, nw);
    cvt_bf16_kernel<<<(nw + 255) / 256, 256, 0, stream>>>(phi_w, pwbf, nw);
    cvt_bf16_kernel<<<(ng + 255) / 256, 256, 0, stream>>>(sup_emb, gxbf, ng);

    dim3 grid(Ndim / 64, Bsz);
    proj_kernel<<<grid, 128, 0, stream>>>(x_query,   twbf, theta_b, thetaX);
    proj_kernel<<<grid, 128, 0, stream>>>(x_support, pwbf, phi_b,   phiTp);

    attn_kernel<<<grid, 128, 0, stream>>>(thetaX, phiTp, gxbf, sup_mask, out);
}
